// MoEMLABlock_14207751815645
// MI455X (gfx1250) — matrix operand order repaired and checked by reading, not compiled, not run on hardware
//
#include <hip/hip_runtime.h>
#include <math.h>

// ---------------- problem constants ----------------
constexpr int HD  = 1024;           // hidden
constexpr int NH  = 16;             // heads
constexpr int DH  = 64;             // head dim
constexpr int LV  = 2;              // attention levels
constexpr int NE  = 8;              // experts
constexpr int FF  = 4096;           // expert hidden
constexpr int BBS = 2;              // batch
constexpr int SS  = 1024;           // seq
constexpr int BS  = BBS * SS;       // tokens = 2048

typedef __attribute__((ext_vector_type(16))) _Float16 v16h;
typedef __attribute__((ext_vector_type(8)))  float    v8f;

union U4H8 { uint4 u; _Float16 h[8]; };

// ---------------- CDNA5 async global->LDS copy (ASYNCcnt-tracked) ----------
// VGLOBAL "GV" form: vdst = LDS byte offset (32-bit VGPR),
//                    vaddr = 64-bit global address (VGPR pair), no SADDR.
__device__ inline unsigned lds_off(const void* p) {
  // flat shared pointers carry the LDS byte offset in addr[31:0]
  return (unsigned)(unsigned long long)p;
}
__device__ inline void async_ld_b128(unsigned lds, const void* g) {
  asm volatile("global_load_async_to_lds_b128 %0, %1, off"
               :: "v"(lds), "v"((unsigned long long)g) : "memory");
}
__device__ inline void wait_async() {
  asm volatile("s_wait_asynccnt 0x0" ::: "memory");
}

// ---------------- helpers ----------------
__device__ inline float redmax16(float v) {
  for (int m = 1; m < 16; m <<= 1) v = fmaxf(v, __shfl_xor(v, m, 32));
  return v;
}
__device__ inline float redsum16(float v) {
  for (int m = 1; m < 16; m <<= 1) v += __shfl_xor(v, m, 32);
  return v;
}
__device__ inline float gelu_tanh(float x) {
  float x3 = x * x * x;
  return 0.5f * x * (1.f + tanhf(0.7978845608028654f * (x + 0.044715f * x3)));
}

// ---------------- fp32 -> f16 convert ----------------
__global__ void k_f32_to_f16(const float* __restrict__ src,
                             _Float16* __restrict__ dst, long n) {
  long i = (long)blockIdx.x * blockDim.x + threadIdx.x;
  long stride = (long)gridDim.x * blockDim.x;
  for (; i < n; i += stride) dst[i] = (_Float16)src[i];
}

// ---------------- layernorm (per token) ----------------
__global__ void k_layernorm_f16(const float* __restrict__ x,
                                const float* __restrict__ g,
                                const float* __restrict__ b,
                                _Float16* __restrict__ y) {
  int t = blockIdx.x, tid = threadIdx.x;
  const float* xr = x + (long)t * HD;
  float s = 0.f, s2 = 0.f;
  for (int h = tid; h < HD; h += 256) { float v = xr[h]; s += v; s2 += v * v; }
  __shared__ float rs[256], rs2[256];
  rs[tid] = s; rs2[tid] = s2; __syncthreads();
  for (int o = 128; o > 0; o >>= 1) {
    if (tid < o) { rs[tid] += rs[tid + o]; rs2[tid] += rs2[tid + o]; }
    __syncthreads();
  }
  float mean = rs[0] / HD;
  float var  = rs2[0] / HD - mean * mean;
  float rstd = rsqrtf(var + 1e-5f);
  for (int h = tid; h < HD; h += 256) {
    float v = (xr[h] - mean) * rstd * g[h] + b[h];
    y[(long)t * HD + h] = (_Float16)v;
  }
}

// ---------------- generic WMMA GEMM ----------------
// C[MxN] = act( A[MxK](f16) x B[KxN](f16) + bias[N] )
// gather: A row = gather[base+m] >> gshift ; scatter: C row = scatter[base+m]
// ebase: device per-expert row offsets (base[e], count = base[e+1]-base[e])
// Tile: BM=128 BN=128 BK=32, 8 waves (4 x 2), wave tile 32x64.
// A tile comes in via global_load_async_to_lds_b128; B tile is transposed
// into LDS manually (async copies cannot transpose).
__global__ void __launch_bounds__(256)
k_gemm_f16(const _Float16* __restrict__ A, long lda,
           const _Float16* __restrict__ Bm, long ldb,
           const float* __restrict__ bias,
           _Float16* __restrict__ Cf16, float* __restrict__ Cf32, long ldc,
           int M, int N, int K,
           const int* __restrict__ gather, int gshift,
           const int* __restrict__ scatter,
           const int* __restrict__ ebase, int eid,
           int act) {
  int base = 0, Mloc = M;
  if (ebase) { base = ebase[eid]; Mloc = ebase[eid + 1] - base; }
  int m0 = blockIdx.y * 128;
  if (m0 >= Mloc) return;
  int n0 = blockIdx.x * 128;

  __shared__ _Float16 As[128 * 32];   // row-major, stride 32 halves
  __shared__ _Float16 Bt[128 * 32];   // [n][k], stride 32 halves

  int tid = threadIdx.x, lane = tid & 31, wid = tid >> 5;
  int wm = wid & 3, wn = wid >> 2;
  int g  = lane >> 4, lm = lane & 15;

  // per-thread LDS destinations for the async A-tile copy
  unsigned asdst[2];
  for (int p = 0; p < 2; p++)
    asdst[p] = lds_off(&As[(p * 64 + (tid >> 2)) * 32 + (tid & 3) * 8]);

  // precompute A row element-offsets for the 2 load passes of this thread
  long arow[2];
  for (int p = 0; p < 2; p++) {
    int ml = p * 64 + (tid >> 2);
    int mm = m0 + ml; if (mm >= Mloc) mm = 0;
    int ar = gather ? (gather[base + mm] >> gshift) : (base + mm);
    arow[p] = (long)ar * lda + (long)((tid & 3) * 8);
  }

  v8f acc[2][4];
  for (int i = 0; i < 2; i++)
    for (int j = 0; j < 4; j++)
      for (int r = 0; r < 8; r++) acc[i][j][r] = 0.f;

  int nk = K / 32;
  for (int kt = 0; kt < nk; kt++) {
    int k0 = kt * 32;
    // A tile: async copy straight into LDS (128 rows x 32 halves)
    for (int p = 0; p < 2; p++)
      async_ld_b128(asdst[p], A + arow[p] + k0);
    // B tile: 32 k-rows x 128 n, stored transposed
    for (int p = 0; p < 2; p++) {
      int kr = p * 16 + (tid >> 4);
      int nc = (tid & 15) * 8;
      U4H8 t; t.u = *(const uint4*)(Bm + (long)(k0 + kr) * ldb + n0 + nc);
      for (int jj = 0; jj < 8; jj++) Bt[(nc + jj) * 32 + kr] = t.h[jj];
    }
    if (k0 + 32 < K)
      __builtin_prefetch(Bm + (long)(k0 + 32) * ldb + n0 + (long)((tid & 15) * 8), 0, 0);
    wait_async();
    __syncthreads();

    v16h afrag[2], bfrag[4];
    for (int im = 0; im < 2; im++) {
      int row = wm * 32 + im * 16 + lm;
      ((uint4*)&afrag[im])[0] = *(const uint4*)&As[row * 32 + g * 8];
      ((uint4*)&afrag[im])[1] = *(const uint4*)&As[row * 32 + 16 + g * 8];
    }
    for (int fn = 0; fn < 4; fn++) {
      int n = wn * 64 + fn * 16 + lm;
      ((uint4*)&bfrag[fn])[0] = *(const uint4*)&Bt[n * 32 + g * 8];
      ((uint4*)&bfrag[fn])[1] = *(const uint4*)&Bt[n * 32 + 16 + g * 8];
    }
    for (int im = 0; im < 2; im++)
      for (int fn = 0; fn < 4; fn++)
        acc[im][fn] = __builtin_amdgcn_wmma_f32_16x16x32_f16(
            false, afrag[im], false, bfrag[fn], (short)0, acc[im][fn], false, false);
    __syncthreads();
  }

  // epilogue
  for (int im = 0; im < 2; im++) {
    for (int fn = 0; fn < 4; fn++) {
      int gcol = n0 + wn * 64 + fn * 16 + lm;
      float bv = bias ? bias[gcol] : 0.f;
      for (int r = 0; r < 8; r++) {
        int ml = wm * 32 + im * 16 + r + g * 8;
        int mm = m0 + ml;
        if (mm < Mloc) {
          float v = acc[im][fn][r] + bv;
          if (act == 1) v = gelu_tanh(v);
          int crow = scatter ? scatter[base + mm] : (base + mm);
          if (Cf16) Cf16[(long)crow * ldc + gcol] = (_Float16)v;
          else      Cf32[(long)crow * ldc + gcol] = v;
        }
      }
    }
  }
}

// ---------------- flash attention (per level/batch/head, 128-row q tile) ----
__global__ void __launch_bounds__(256)
k_attention(const _Float16* __restrict__ Qg, const _Float16* __restrict__ Kg,
            const _Float16* __restrict__ Vg, const int* __restrict__ mask,
            _Float16* __restrict__ Og) {
  int qt = blockIdx.x, head = blockIdx.y, z = blockIdx.z;
  int lv = z >> 1, b = z & 1;
  long base = ((long)lv * BS + (long)b * SS) * HD + (long)head * DH;

  __shared__ _Float16 Qs[128 * 64];     // [qrow][d]
  __shared__ _Float16 Ks[64 * 64];      // [key][d]
  __shared__ _Float16 Vt[64 * 64];      // [d][key]
  __shared__ _Float16 Ps[8][16 * 64];   // per-wave P tile [qrow16][key64]
  __shared__ float    maskb[64];

  int tid = threadIdx.x, lane = tid & 31, wid = tid >> 5;
  int g = lane >> 4, lm = lane & 15;
  int q0 = qt * 128;

  // Q tile: async copy into LDS (waited before first use below)
  for (int p = 0; p < 4; p++) {
    int r = p * 32 + (tid >> 3); int c = (tid & 7) * 8;
    async_ld_b128(lds_off(&Qs[r * 64 + c]),
                  Qg + base + (long)(q0 + r) * HD + c);
  }

  float mrow[8], lrow[8];
  v8f cacc[4];
  for (int r = 0; r < 8; r++) { mrow[r] = -1e30f; lrow[r] = 0.f; }
  for (int f = 0; f < 4; f++)
    for (int r = 0; r < 8; r++) cacc[f][r] = 0.f;

  for (int kb = 0; kb < SS / 64; kb++) {
    int key0 = kb * 64;
    __syncthreads();  // previous iteration's LDS reads complete
    for (int p = 0; p < 2; p++) {
      int r = p * 32 + (tid >> 3); int c = (tid & 7) * 8;
      // K tile row-major: async straight to LDS
      async_ld_b128(lds_off(&Ks[r * 64 + c]),
                    Kg + base + (long)(key0 + r) * HD + c);
      // V tile needs transpose: manual path
      U4H8 t; t.u = *(const uint4*)(Vg + base + (long)(key0 + r) * HD + c);
      for (int jj = 0; jj < 8; jj++) Vt[(c + jj) * 64 + r] = t.h[jj];
    }
    if (tid < 64)
      maskb[tid] = (1.f - (float)mask[b * SS + key0 + tid]) * -1e9f;
    wait_async();   // covers Q (first iteration) and this K tile
    __syncthreads();

    // scores S = Q K^T / sqrt(DH) + bias
    v8f s[4];
    for (int f = 0; f < 4; f++)
      for (int r = 0; r < 8; r++) s[f][r] = 0.f;
    for (int ks = 0; ks < 2; ks++) {
      v16h a;
      int row = wid * 16 + lm;
      ((uint4*)&a)[0] = *(const uint4*)&Qs[row * 64 + ks * 32 + g * 8];
      ((uint4*)&a)[1] = *(const uint4*)&Qs[row * 64 + ks * 32 + 16 + g * 8];
      for (int f = 0; f < 4; f++) {
        v16h bb;
        int n = f * 16 + lm;
        ((uint4*)&bb)[0] = *(const uint4*)&Ks[n * 64 + ks * 32 + g * 8];
        ((uint4*)&bb)[1] = *(const uint4*)&Ks[n * 64 + ks * 32 + 16 + g * 8];
        s[f] = __builtin_amdgcn_wmma_f32_16x16x32_f16(
            false, a, false, bb, (short)0, s[f], false, false);
      }
    }
    for (int f = 0; f < 4; f++) {
      float mb = maskb[f * 16 + lm];
      for (int r = 0; r < 8; r++) s[f][r] = s[f][r] * 0.125f + mb;
    }
    // online softmax
    float nm[8], rs[8];
    for (int r = 0; r < 8; r++) {
      float v = -1e30f;
      for (int f = 0; f < 4; f++) v = fmaxf(v, s[f][r]);
      v = redmax16(v);
      nm[r] = fmaxf(mrow[r], v);
      rs[r] = 0.f;
    }
    for (int f = 0; f < 4; f++)
      for (int r = 0; r < 8; r++) {
        float p = __expf(s[f][r] - nm[r]);
        s[f][r] = p; rs[r] += p;
      }
    for (int r = 0; r < 8; r++) {
      rs[r] = redsum16(rs[r]);
      float sc = __expf(mrow[r] - nm[r]);
      lrow[r] = lrow[r] * sc + rs[r];
      mrow[r] = nm[r];
      for (int f = 0; f < 4; f++) cacc[f][r] *= sc;
    }
    // stage P to per-wave LDS in A-fragment-friendly row-major layout
    for (int f = 0; f < 4; f++)
      for (int r = 0; r < 8; r++)
        Ps[wid][(r + g * 8) * 64 + f * 16 + lm] = (_Float16)s[f][r];
    __syncthreads();

    // ctx += P @ V
    for (int ks = 0; ks < 2; ks++) {
      v16h a;
      ((uint4*)&a)[0] = *(const uint4*)&Ps[wid][lm * 64 + ks * 32 + g * 8];
      ((uint4*)&a)[1] = *(const uint4*)&Ps[wid][lm * 64 + ks * 32 + 16 + g * 8];
      for (int fd = 0; fd < 4; fd++) {
        v16h bb;
        int n = fd * 16 + lm;
        ((uint4*)&bb)[0] = *(const uint4*)&Vt[n * 64 + ks * 32 + g * 8];
        ((uint4*)&bb)[1] = *(const uint4*)&Vt[n * 64 + ks * 32 + 16 + g * 8];
        cacc[fd] = __builtin_amdgcn_wmma_f32_16x16x32_f16(
            false, a, false, bb, (short)0, cacc[fd], false, false);
      }
    }
  }

  // normalize + store ctx (f16)
  for (int fd = 0; fd < 4; fd++)
    for (int r = 0; r < 8; r++) {
      int srow = q0 + wid * 16 + r + g * 8;
      int col  = fd * 16 + lm;
      float v = cacc[fd][r] / lrow[r];
      Og[base + (long)srow * HD + col] = (_Float16)v;
    }
}

// ---------------- level-softmax combine + residual 1 ----------------
__global__ void k_combine_levels(const float* __restrict__ hs,
                                 const float* __restrict__ olvl,
                                 const float* __restrict__ lvl,
                                 float* __restrict__ xout) {
  long i = (long)blockIdx.x * 256 + threadIdx.x;
  float a = lvl[0], c = lvl[1];
  float m = fmaxf(a, c);
  float e0 = __expf(a - m), e1 = __expf(c - m);
  float inv = 1.f / (e0 + e1);
  xout[i] = hs[i] + (e0 * inv) * olvl[i] + (e1 * inv) * olvl[(long)BS * HD + i];
}

// ---------------- router ----------------
__global__ void k_zero_i32(int* p, int n) {
  int i = blockIdx.x * blockDim.x + threadIdx.x;
  if (i < n) p[i] = 0;
}

__global__ void k_router(const _Float16* __restrict__ xln2,
                         const float* __restrict__ Wr, const float* __restrict__ br,
                         float* __restrict__ gates, int* __restrict__ ei,
                         int* __restrict__ counts) {
  int t = blockIdx.x, tid = threadIdx.x;
  float p[NE];
  for (int e = 0; e < NE; e++) p[e] = 0.f;
  for (int h = tid; h < HD; h += 256) {
    float xv = (float)xln2[(long)t * HD + h];
    for (int e = 0; e < NE; e++) p[e] += xv * Wr[h * NE + e];
  }
  __shared__ float red[NE * 256];
  for (int e = 0; e < NE; e++) red[e * 256 + tid] = p[e];
  __syncthreads();
  for (int o = 128; o > 0; o >>= 1) {
    if (tid < o)
      for (int e = 0; e < NE; e++) red[e * 256 + tid] += red[e * 256 + tid + o];
    __syncthreads();
  }
  if (tid == 0) {
    float lg[NE], pr[NE], mx = -1e30f, sum = 0.f;
    for (int e = 0; e < NE; e++) { lg[e] = red[e * 256] + br[e]; mx = fmaxf(mx, lg[e]); }
    for (int e = 0; e < NE; e++) { pr[e] = __expf(lg[e] - mx); sum += pr[e]; }
    for (int e = 0; e < NE; e++) pr[e] /= sum;
    int i1 = 0;
    for (int e = 1; e < NE; e++) if (pr[e] > pr[i1]) i1 = e;
    int i2 = (i1 == 0) ? 1 : 0;
    for (int e = 0; e < NE; e++) if (e != i1 && pr[e] > pr[i2]) i2 = e;
    float s2 = pr[i1] + pr[i2];
    gates[t * 2 + 0] = pr[i1] / s2;
    gates[t * 2 + 1] = pr[i2] / s2;
    ei[t * 2 + 0] = i1; ei[t * 2 + 1] = i2;
    atomicAdd(&counts[i1], 1);
    atomicAdd(&counts[i2], 1);
  }
}

__global__ void k_offsets(const int* __restrict__ counts, int* __restrict__ basep,
                          int* __restrict__ cursor) {
  int acc = 0;
  for (int e = 0; e < NE; e++) { basep[e] = acc; cursor[e] = acc; acc += counts[e]; }
  basep[NE] = acc;
}

__global__ void k_assign(const int* __restrict__ ei, int* __restrict__ cursor,
                         int* __restrict__ perm) {
  int t = blockIdx.x * 256 + threadIdx.x;
  if (t >= BS) return;
  for (int k = 0; k < 2; k++) {
    int e = ei[t * 2 + k];
    int pos = atomicAdd(&cursor[e], 1);
    perm[pos] = t * 2 + k;
  }
}

// ---------------- final combine: residual 2 ----------------
__global__ void k_final(const float* __restrict__ x, const float* __restrict__ yk,
                        const float* __restrict__ gates, float* __restrict__ out) {
  long i = (long)blockIdx.x * 256 + threadIdx.x;
  long t = i >> 10;           // / HD
  int  c = (int)(i & 1023);
  out[i] = x[i] + gates[t * 2 + 0] * yk[(t * 2 + 0) * HD + c]
                + gates[t * 2 + 1] * yk[(t * 2 + 1) * HD + c];
}

// ============================================================================
extern "C" void kernel_launch(void* const* d_in, const int* in_sizes, int n_in,
                              void* d_out, int out_size, void* d_ws, size_t ws_size,
                              hipStream_t stream) {
  const float* hs   = (const float*)d_in[0];
  const int*   am   = (const int*)  d_in[1];
  const float* ln1g = (const float*)d_in[2];
  const float* ln1b = (const float*)d_in[3];
  const float* ln2g = (const float*)d_in[4];
  const float* ln2b = (const float*)d_in[5];
  const float* Wq   = (const float*)d_in[6];
  const float* Wk   = (const float*)d_in[7];
  const float* Wv   = (const float*)d_in[8];
  const float* Wo   = (const float*)d_in[9];
  const float* bq   = (const float*)d_in[10];
  const float* bk   = (const float*)d_in[11];
  const float* bv   = (const float*)d_in[12];
  const float* bo   = (const float*)d_in[13];
  const float* lvl  = (const float*)d_in[14];
  const float* Wr   = (const float*)d_in[15];
  const float* br   = (const float*)d_in[16];
  const float* W1   = (const float*)d_in[17];
  const float* b1   = (const float*)d_in[18];
  const float* W2   = (const float*)d_in[19];
  const float* b2   = (const float*)d_in[20];

  // ---- workspace carve-up ----
  char* w = (char*)d_ws;
  auto alloc = [&](size_t bytes) {
    void* p = (void*)w;
    w += (bytes + 255) & ~(size_t)255;
    return p;
  };
  _Float16* Wq16  = (_Float16*)alloc((size_t)LV * HD * HD * 2);
  _Float16* Wk16  = (_Float16*)alloc((size_t)LV * HD * HD * 2);
  _Float16* Wv16  = (_Float16*)alloc((size_t)LV * HD * HD * 2);
  _Float16* Wo16  = (_Float16*)alloc((size_t)LV * HD * HD * 2);
  _Float16* W116  = (_Float16*)alloc((size_t)NE * HD * FF * 2);
  _Float16* W216  = (_Float16*)alloc((size_t)NE * FF * HD * 2);
  _Float16* XLN1  = (_Float16*)alloc((size_t)BS * HD * 2);
  _Float16* Q16   = (_Float16*)alloc((size_t)LV * BS * HD * 2);
  _Float16* K16   = (_Float16*)alloc((size_t)LV * BS * HD * 2);
  _Float16* V16   = (_Float16*)alloc((size_t)LV * BS * HD * 2);
  _Float16* CTX16 = (_Float16*)alloc((size_t)LV * BS * HD * 2);
  float*    OLVL  = (float*)   alloc((size_t)LV * BS * HD * 4);
  float*    XBUF  = (float*)   alloc((size_t)BS * HD * 4);
  _Float16* XLN2  = (_Float16*)alloc((size_t)BS * HD * 2);
  float*    GATES = (float*)   alloc((size_t)BS * 2 * 4);
  int*      EI    = (int*)     alloc((size_t)BS * 2 * 4);
  int*      CNT   = (int*)     alloc((size_t)NE * 4);
  int*      BASE  = (int*)     alloc((size_t)(NE + 1) * 4);
  int*      CUR   = (int*)     alloc((size_t)NE * 4);
  int*      PERM  = (int*)     alloc((size_t)BS * 2 * 4);
  _Float16* HBUF  = (_Float16*)alloc((size_t)BS * 2 * FF * 2);
  float*    YK    = (float*)   alloc((size_t)BS * 2 * HD * 4);

  // ---- 1. weight conversion to f16 ----
  k_f32_to_f16<<<4096, 256, 0, stream>>>(Wq, Wq16, (long)LV * HD * HD);
  k_f32_to_f16<<<4096, 256, 0, stream>>>(Wk, Wk16, (long)LV * HD * HD);
  k_f32_to_f16<<<4096, 256, 0, stream>>>(Wv, Wv16, (long)LV * HD * HD);
  k_f32_to_f16<<<4096, 256, 0, stream>>>(Wo, Wo16, (long)LV * HD * HD);
  k_f32_to_f16<<<8192, 256, 0, stream>>>(W1, W116, (long)NE * HD * FF);
  k_f32_to_f16<<<8192, 256, 0, stream>>>(W2, W216, (long)NE * FF * HD);

  // ---- 2. LN1 ----
  k_layernorm_f16<<<BS, 256, 0, stream>>>(hs, ln1g, ln1b, XLN1);

  // ---- 3. QKV projections per level ----
  for (int l = 0; l < LV; l++) {
    k_gemm_f16<<<dim3(HD / 128, BS / 128), 256, 0, stream>>>(
        XLN1, HD, Wq16 + (long)l * HD * HD, HD, bq + l * HD,
        Q16 + (long)l * BS * HD, nullptr, HD, BS, HD, HD,
        nullptr, 0, nullptr, nullptr, 0, 0);
    k_gemm_f16<<<dim3(HD / 128, BS / 128), 256, 0, stream>>>(
        XLN1, HD, Wk16 + (long)l * HD * HD, HD, bk + l * HD,
        K16 + (long)l * BS * HD, nullptr, HD, BS, HD, HD,
        nullptr, 0, nullptr, nullptr, 0, 0);
    k_gemm_f16<<<dim3(HD / 128, BS / 128), 256, 0, stream>>>(
        XLN1, HD, Wv16 + (long)l * HD * HD, HD, bv + l * HD,
        V16 + (long)l * BS * HD, nullptr, HD, BS, HD, HD,
        nullptr, 0, nullptr, nullptr, 0, 0);
  }

  // ---- 4. attention ----
  k_attention<<<dim3(SS / 128, NH, LV * BBS), 256, 0, stream>>>(
      Q16, K16, V16, am, CTX16);

  // ---- 5. output projection per level (f32), combine + residual ----
  for (int l = 0; l < LV; l++) {
    k_gemm_f16<<<dim3(HD / 128, BS / 128), 256, 0, stream>>>(
        CTX16 + (long)l * BS * HD, HD, Wo16 + (long)l * HD * HD, HD, bo + l * HD,
        nullptr, OLVL + (long)l * BS * HD, HD, BS, HD, HD,
        nullptr, 0, nullptr, nullptr, 0, 0);
  }
  k_combine_levels<<<(BS * HD) / 256, 256, 0, stream>>>(hs, OLVL, lvl, XBUF);

  // ---- 6. LN2 + router ----
  k_layernorm_f16<<<BS, 256, 0, stream>>>(XBUF, ln2g, ln2b, XLN2);
  k_zero_i32<<<1, 64, 0, stream>>>(CNT, NE);
  k_router<<<BS, 256, 0, stream>>>(XLN2, Wr, br, GATES, EI, CNT);
  k_offsets<<<1, 1, 0, stream>>>(CNT, BASE, CUR);
  k_assign<<<BS / 256, 256, 0, stream>>>(EI, CUR, PERM);

  // ---- 7. MoE grouped GEMMs ----
  for (int e = 0; e < NE; e++) {
    // h = gelu(Xg @ W1[e] + b1[e])  -> HBUF rows [base[e], base[e]+cnt)
    k_gemm_f16<<<dim3(FF / 128, BS / 128), 256, 0, stream>>>(
        XLN2, HD, W116 + (long)e * HD * FF, FF, b1 + (long)e * FF,
        HBUF, nullptr, FF, BS * 2, FF, HD,
        PERM, 1, nullptr, BASE, e, 1);
    // y = h @ W2[e] + b2[e] -> scatter rows to YK[2t+k]
    k_gemm_f16<<<dim3(HD / 128, BS / 128), 256, 0, stream>>>(
        HBUF, FF, W216 + (long)e * FF * HD, HD, b2 + (long)e * HD,
        nullptr, YK, HD, BS * 2, HD, FF,
        nullptr, 0, PERM, BASE, e, 0);
  }

  // ---- 8. final combine + residual 2 ----
  k_final<<<(BS * HD) / 256, 256, 0, stream>>>(XBUF, YK, GATES, (float*)d_out);

  (void)in_sizes; (void)n_in; (void)out_size; (void)ws_size;
}
